// AttentionNestedNERModel_31241592111113
// MI455X (gfx1250) — matrix operand order repaired and checked by reading, not compiled, not run on hardware
//
#include <hip/hip_runtime.h>

// ---------------------------------------------------------------------------
// Types
// ---------------------------------------------------------------------------
typedef __bf16 bf16_t;
typedef __attribute__((ext_vector_type(8)))  bf16_t v8bf;
typedef __attribute__((ext_vector_type(16))) bf16_t v16bf;
typedef __attribute__((ext_vector_type(8)))  float  v8f;

#define DEV __device__ __forceinline__

// Model constants (match reference)
constexpr int S_  = 128;
constexpr int B_  = 32;
constexpr int E_  = 512;
constexpr int H_  = 512;
constexpr int DH_ = 1024;
constexpr int TAG_ = 11;
constexpr int L_  = 3;

// ---------------------------------------------------------------------------
// Workspace layout (bytes)
// ---------------------------------------------------------------------------
constexpr size_t OFF_CTRL   = 0;                                   // 256 B barrier ctrl
constexpr size_t OFF_STS    = 256;                                 // dec carry s  [32,1024] f32
constexpr size_t OFF_STC    = OFF_STS    + (size_t)B_*DH_*4;       // dec carry c
constexpr size_t OFF_STBF   = OFF_STC    + (size_t)B_*DH_*4;       // dec state bf16
constexpr size_t OFF_ENCS   = OFF_STBF   + (size_t)B_*DH_*2;       // enc s [2][32,512] f32
constexpr size_t OFF_ENCC   = OFF_ENCS   + (size_t)2*B_*H_*4;      // enc c
constexpr size_t OFF_ENCSBF = OFF_ENCC   + (size_t)2*B_*H_*4;      // enc s bf16
constexpr size_t OFF_ZEND   = OFF_ENCSBF + (size_t)2*B_*H_*2;      // end of zero-init region
constexpr size_t OFF_LOG    = OFF_ZEND;                            // logits [32,128] f32
constexpr size_t OFF_ATT    = OFF_LOG    + (size_t)B_*S_*4;        // softmax weights
constexpr size_t OFF_CTX    = OFF_ATT    + (size_t)B_*S_*4;        // ctx bf16 [32,1024]
constexpr size_t OFF_GATES  = OFF_CTX    + (size_t)B_*DH_*2;       // gates scratch [32,4096] f32
constexpr size_t OFF_XE     = OFF_GATES  + (size_t)B_*4*DH_*4;     // embedded input bf16 [S*B,E]
constexpr size_t OFF_WFIH   = OFF_XE     + (size_t)S_*B_*E_*2;     // enc_f_Wih bf16 [2048,512]
constexpr size_t OFF_WFHH   = OFF_WFIH   + (size_t)4*H_*E_*2;      // enc_f_Whh bf16 [2048,512]
constexpr size_t OFF_WBIH   = OFF_WFHH   + (size_t)4*H_*H_*2;
constexpr size_t OFF_WBHH   = OFF_WBIH   + (size_t)4*H_*E_*2;
constexpr size_t OFF_WDIH   = OFF_WBHH   + (size_t)4*H_*H_*2;      // dec_Wih bf16 [4096,3072]
constexpr size_t OFF_WDHH   = OFF_WDIH   + (size_t)4*DH_*3*DH_*2;  // dec_Whh bf16 [4096,1024]
constexpr size_t OFF_WATT   = OFF_WDHH   + (size_t)4*DH_*DH_*2;    // attn_W bf16 [1024,1024]
constexpr size_t OFF_GF     = OFF_WATT   + (size_t)DH_*DH_*2;      // enc fwd input gates f32 [4096,2048]
constexpr size_t OFF_GB     = OFF_GF     + (size_t)S_*B_*4*H_*4;
constexpr size_t OFF_HBF    = OFF_GB     + (size_t)S_*B_*4*H_*4;   // h bf16 [S*B,1024]
constexpr size_t OFF_HW     = OFF_HBF    + (size_t)S_*B_*DH_*2;    // hW bf16 [S*B,1024]
constexpr size_t OFF_YS0    = OFF_HW     + (size_t)S_*B_*DH_*2;    // level-0 ys bf16
constexpr size_t OFF_YSF    = OFF_YS0    + (size_t)S_*B_*DH_*2;    // ys f32 [3][S*B,1024]
constexpr size_t OFF_P      = OFF_YSF    + (size_t)L_*S_*B_*DH_*4; // static decoder gates f32 [4096,4096]
constexpr size_t WS_NEED    = OFF_P      + (size_t)S_*B_*4*DH_*4;

// ---------------------------------------------------------------------------
// Scalar helpers
// ---------------------------------------------------------------------------
DEV float  bf2f(bf16_t x) { return (float)x; }
DEV bf16_t f2bf(float  x) { return (bf16_t)x; }

DEV float sigmoidf_(float x) { return 1.f / (1.f + __expf(-x)); }
DEV float tanhf_(float x) {
  x = fminf(fmaxf(x, -15.f), 15.f);
  float e = __expf(2.f * x);
  return (e - 1.f) / (e + 1.f);
}

// ---------------------------------------------------------------------------
// WMMA helpers: bf16 16x16x32, f32 accumulate.
// Fragments: A 16x32, B 32x16, k = 16*(lane/16) + s (not ISA table order); C/D per CDNA5 ISA 7.12.2.
// A (activations, row-major [M,K]) and W (weights, row-major [N,K]) both give
// K-contiguous 16B chunks per lane -> global_load_b128 / ds_load_b128.
// ---------------------------------------------------------------------------
DEV v8f wmma_bf16(v16bf a, v16bf b, v8f c) {
  return __builtin_amdgcn_wmma_f32_16x16x32_bf16(false, a, false, b, (short)0, c,
                                                 false, false);
}

DEV v16bf load_afrag(const bf16_t* __restrict__ A, int lda, int row0, int kb, int lane) {
  int m   = row0 + (lane & 15);
  int off = (lane >> 4) * 16;                 // lanes 16-31 hold K = 16..31
  const bf16_t* base = A + (size_t)m * lda + kb + off;
  v8bf lo = *(const v8bf*)(base);             // K = kb+off   .. +7
  v8bf hi = *(const v8bf*)(base + 8);         // K = kb+off+8 .. +7
  return __builtin_shufflevector(lo, hi, 0,1,2,3,4,5,6,7,8,9,10,11,12,13,14,15);
}

DEV v16bf load_bfrag(const bf16_t* __restrict__ W, int ldw, int col0, int kb, int lane) {
  int n  = col0 + (lane & 15);
  int kh = (lane >> 4) * 16;                  // lanes 16-31 hold K = 16..31
  const bf16_t* base = W + (size_t)n * ldw + kb + kh;
  v8bf lo = *(const v8bf*)(base);
  v8bf hi = *(const v8bf*)(base + 8);
  return __builtin_shufflevector(lo, hi, 0,1,2,3,4,5,6,7,8,9,10,11,12,13,14,15);
}

// B fragment from an LDS-resident weight slice (ld = K, rows are local cols).
DEV v16bf load_bfrag_lds(const bf16_t* Ws, int ldw, int nLocal, int kb, int lane) {
  int n  = nLocal + (lane & 15);
  int kh = (lane >> 4) * 16;
  const bf16_t* base = Ws + n * ldw + kb + kh;
  v8bf lo = *(const v8bf*)(base);
  v8bf hi = *(const v8bf*)(base + 8);
  return __builtin_shufflevector(lo, hi, 0,1,2,3,4,5,6,7,8,9,10,11,12,13,14,15);
}

// ---------------------------------------------------------------------------
// Device-wide sense-reversing barrier (persistent cooperative kernels).
// ctrl[0] = arrival counter (0 at kernel entry/exit), ctrl[1] = generation.
// ---------------------------------------------------------------------------
DEV void grid_barrier(unsigned* ctrl, unsigned nblk) {
  __threadfence();
  __syncthreads();
  if (threadIdx.x == 0) {
    volatile unsigned* gen = (volatile unsigned*)(ctrl + 1);
    unsigned g = *gen;
    unsigned arrived = atomicAdd(ctrl, 1u);
    if (arrived == nblk - 1u) {
      atomicExch(ctrl, 0u);
      __threadfence();
      atomicAdd(ctrl + 1, 1u);
    } else {
      while (*gen == g) { __builtin_amdgcn_s_sleep(1); }
    }
    __threadfence();
  }
  __syncthreads();
}

// ---------------------------------------------------------------------------
// Small utility kernels
// ---------------------------------------------------------------------------
__global__ void zero_kernel(unsigned* p, int nwords) {
  for (int i = blockIdx.x * blockDim.x + threadIdx.x; i < nwords;
       i += gridDim.x * blockDim.x)
    p[i] = 0u;
}

__global__ void f2bf_kernel(const float* __restrict__ src, bf16_t* __restrict__ dst, int n) {
  for (int i = blockIdx.x * blockDim.x + threadIdx.x; i < n;
       i += gridDim.x * blockDim.x)
    dst[i] = f2bf(src[i]);
}

// x[s,b,:] = emb[seqs[b,s]] -> bf16, row index = s*B + b
__global__ void embed_kernel(const int* __restrict__ seqs, const float* __restrict__ emb,
                             bf16_t* __restrict__ xe) {
  int sb = blockIdx.x;                 // s*B + b
  int s = sb >> 5, b = sb & 31;
  int tok = seqs[b * S_ + s];
  const float* row = emb + (size_t)tok * E_;
  bf16_t* dst = xe + (size_t)sb * E_;
  for (int e = threadIdx.x; e < E_; e += blockDim.x) dst[e] = f2bf(row[e]);
}

// ---------------------------------------------------------------------------
// Batched WMMA GEMM:  C[M,N] = A1@W1^T (+ A2@W2^T) (+ bias)
// A row-major [M,K] bf16, W row-major [N,K] bf16.
// Block tile 32(M) x 128(N); 8 waves, wave w -> 16 N-cols, 2 M-tiles.
// grid = (N/128, M/32)
// ---------------------------------------------------------------------------
__global__ void __launch_bounds__(256)
wmma_gemm_dual(const bf16_t* __restrict__ A1, const bf16_t* __restrict__ W1,
               int K1, int lda1, int ldw1,
               const bf16_t* __restrict__ A2, const bf16_t* __restrict__ W2,
               int K2, int lda2, int ldw2,
               const float* __restrict__ bias,
               float* __restrict__ Cf, bf16_t* __restrict__ Cb, int N) {
  int lane = threadIdx.x & 31, wv = threadIdx.x >> 5;
  int col0 = blockIdx.x * 128 + wv * 16;
  int row0 = blockIdx.y * 32;

  v8f acc0 = {}; v8f acc1 = {};
  for (int kb = 0; kb < K1; kb += 32) {
    v16bf bfr = load_bfrag(W1, ldw1, col0, kb, lane);
    v16bf a0  = load_afrag(A1, lda1, row0,      kb, lane);
    v16bf a1  = load_afrag(A1, lda1, row0 + 16, kb, lane);
    acc0 = wmma_bf16(a0, bfr, acc0);
    acc1 = wmma_bf16(a1, bfr, acc1);
  }
  if (A2) {
    for (int kb = 0; kb < K2; kb += 32) {
      v16bf bfr = load_bfrag(W2, ldw2, col0, kb, lane);
      v16bf a0  = load_afrag(A2, lda2, row0,      kb, lane);
      v16bf a1  = load_afrag(A2, lda2, row0 + 16, kb, lane);
      acc0 = wmma_bf16(a0, bfr, acc0);
      acc1 = wmma_bf16(a1, bfr, acc1);
    }
  }
  int col = col0 + (lane & 15);
  float bv = bias ? bias[col] : 0.f;
  int rb = row0 + ((lane >> 4) * 8);
#pragma unroll
  for (int r = 0; r < 8; r++) {
    float v0 = acc0[r] + bv;
    float v1 = acc1[r] + bv;
    size_t i0 = (size_t)(rb + r) * N + col;
    size_t i1 = (size_t)(rb + r + 16) * N + col;
    if (Cf) { Cf[i0] = v0; Cf[i1] = v1; }
    if (Cb) { Cb[i0] = f2bf(v0); Cb[i1] = f2bf(v1); }
  }
}

// ---------------------------------------------------------------------------
// Persistent bidirectional encoder. 32 blocks x 256 threads.
// blocks 0-15: forward dir, 16-31: backward dir; each block owns 128 N-cols.
// The block's recurrent-weight slice (128 cols x 512 K bf16 = 128 KB) is
// staged into LDS once; all 128 steps read B-fragments via ds_load_b128.
// ---------------------------------------------------------------------------
__global__ void __launch_bounds__(256)
encoder_kernel(const bf16_t* __restrict__ WhhF, const bf16_t* __restrict__ WhhB,
               const float* __restrict__ Gf, const float* __restrict__ Gb,
               float* __restrict__ encS, float* __restrict__ encC,
               bf16_t* __restrict__ encSbf, float* __restrict__ gatesScr,
               bf16_t* __restrict__ h, unsigned* ctrl) {
  const unsigned NBLK = 32;
  __shared__ bf16_t sW[128 * 512];              // 128 KB weight slice
  int lane = threadIdx.x & 31, wv = threadIdx.x >> 5;
  int dir  = blockIdx.x >> 4;
  int blkN = blockIdx.x & 15;

  const bf16_t* Whh = dir ? WhhB : WhhF;        // [2048, 512]
  const float*  G   = dir ? Gb : Gf;            // [S*B, 2048]
  float*  sS  = encS   + dir * B_ * H_;
  float*  sC  = encC   + dir * B_ * H_;
  bf16_t* sBF = encSbf + dir * B_ * H_;
  float*  gD  = gatesScr + dir * B_ * 4 * H_;   // [32, 2048]

  // stage weight slice: rows [blkN*128, +128) of Whh -> LDS
  {
    const bf16_t* src = Whh + (size_t)(blkN * 128) * H_;
    for (int i = threadIdx.x; i < 128 * 512 / 8; i += 256) {
      *(v8bf*)&sW[i * 8] = *(const v8bf*)&src[i * 8];
    }
  }
  __syncthreads();

  for (int t = 0; t < S_; t++) {
    int sidx = dir ? (S_ - 1 - t) : t;
    // ---- recurrent GEMM: state[32,512] @ Whh[2048,512]^T, cols [blkN*128,+128)
    {
      v8f acc0 = {}; v8f acc1 = {};
      for (int kb = 0; kb < H_; kb += 32) {
        v16bf bfr = load_bfrag_lds(sW, H_, wv * 16, kb, lane);
        v16bf a0  = load_afrag(sBF, H_, 0,  kb, lane);
        v16bf a1  = load_afrag(sBF, H_, 16, kb, lane);
        acc0 = wmma_bf16(a0, bfr, acc0);
        acc1 = wmma_bf16(a1, bfr, acc1);
      }
      int col = blkN * 128 + wv * 16 + (lane & 15);
      int rb  = (lane >> 4) * 8;
      const float* Grow = G + (size_t)(sidx * B_) * (4 * H_);
#pragma unroll
      for (int r = 0; r < 8; r++) {
        int m0 = rb + r, m1 = m0 + 16;
        gD[m0 * 2048 + col] = acc0[r] + Grow[(size_t)m0 * 2048 + col];
        gD[m1 * 2048 + col] = acc1[r] + Grow[(size_t)m1 * 2048 + col];
      }
    }
    grid_barrier(ctrl, NBLK);
    // ---- LSTM cell: 32*512 elems per dir over 16 blocks * 256 threads
    {
      int tloc = blkN * 256 + threadIdx.x;      // 0..4095
#pragma unroll
      for (int e = 0; e < 4; e++) {
        int j = tloc * 4 + e;                   // 0..16383
        int b = j >> 9, d = j & 511;
        float gi = gD[b * 2048 + d];
        float gf = gD[b * 2048 + 512 + d];
        float gg = gD[b * 2048 + 1024 + d];
        float go = gD[b * 2048 + 1536 + d];
        float c  = sC[b * H_ + d];
        float c2 = sigmoidf_(gf) * c + sigmoidf_(gi) * tanhf_(gg);
        float s2 = sigmoidf_(go) * tanhf_(c2);
        sC[b * H_ + d] = c2;
        sS[b * H_ + d] = s2;
        sBF[b * H_ + d] = f2bf(s2);
        h[(size_t)(sidx * B_ + b) * DH_ + dir * H_ + d] = f2bf(s2);
      }
    }
    grid_barrier(ctrl, NBLK);
  }
}

// ---------------------------------------------------------------------------
// Persistent attention decoder, one level. 64 blocks x 256 threads.
// Each block stages its gate-weight slice (64 cols x 1024 K for Wih_ctx and
// for Whh = 256 KB) into WGP LDS once; all 128 steps hit LDS only.
// Per step: logits -> softmax -> ctx -> gates GEMM (WMMA) -> LSTM cell.
// ---------------------------------------------------------------------------
__global__ void __launch_bounds__(256)
decoder_kernel(const bf16_t* __restrict__ hW, const bf16_t* __restrict__ hbf,
               const bf16_t* __restrict__ WihCtx /* dec_Wih cols 0:1024, ld=3072 */,
               const bf16_t* __restrict__ Whh    /* [4096,1024] */,
               const float* __restrict__ P       /* [S*B,4096] static gates+bias */,
               float* __restrict__ stS, float* __restrict__ stC, bf16_t* __restrict__ stBF,
               float* __restrict__ logits, float* __restrict__ attw,
               bf16_t* __restrict__ ctxBF, float* __restrict__ gates,
               float* __restrict__ ysF, bf16_t* __restrict__ ys0BF, int lvl,
               unsigned* ctrl) {
  const unsigned NBLK = 64;
  __shared__ bf16_t sW[2 * 64 * 1024];          // 256 KB: [Wih_ctx | Whh] slices
  int lane  = threadIdx.x & 31, wv = threadIdx.x >> 5;
  int gtid  = blockIdx.x * 256 + threadIdx.x;   // 0..16383
  int gwave = gtid >> 5;                        // 0..511
  int colBase = blockIdx.x * 64;

  // stage both weight slices to LDS
  for (int i = threadIdx.x; i < 64 * 1024 / 8; i += 256) {
    int n = (i * 8) >> 10;
    int k = (i * 8) & 1023;
    *(v8bf*)&sW[(size_t)n * 1024 + k] =
        *(const v8bf*)&WihCtx[(size_t)(colBase + n) * (3 * DH_) + k];
    *(v8bf*)&sW[(size_t)(64 + n) * 1024 + k] =
        *(const v8bf*)&Whh[(size_t)(colBase + n) * DH_ + k];
  }
  __syncthreads();

  for (int t = 0; t < S_; t++) {
    // ---- phase a: logits[b][s'] = <hW[s',b,:], state[b,:]> ; 4096 dots of 1024
    for (int p = gwave; p < B_ * S_; p += 512) {
      int b = p >> 7, sp = p & 127;
      const bf16_t* hwrow = hW + (size_t)(sp * B_ + b) * DH_;
      const bf16_t* st    = stBF + b * DH_;
      float acc = 0.f;
      for (int k = lane; k < DH_; k += 32)
        acc += bf2f(hwrow[k]) * bf2f(st[k]);
      for (int m = 16; m; m >>= 1) acc += __shfl_xor(acc, m, 32);
      if (lane == 0) logits[p] = acc;
    }
    grid_barrier(ctrl, NBLK);
    // ---- phase b: softmax over s' (rows b handled by first 32 waves)
    if (gwave < B_) {
      int b = gwave;
      float v[4];
      float mx = -1e30f;
#pragma unroll
      for (int i = 0; i < 4; i++) {
        v[i] = logits[b * S_ + lane * 4 + i];
        mx = fmaxf(mx, v[i]);
      }
      for (int m = 16; m; m >>= 1) mx = fmaxf(mx, __shfl_xor(mx, m, 32));
      float sum = 0.f;
#pragma unroll
      for (int i = 0; i < 4; i++) { v[i] = __expf(v[i] - mx); sum += v[i]; }
      for (int m = 16; m; m >>= 1) sum += __shfl_xor(sum, m, 32);
      float inv = 1.f / sum;
#pragma unroll
      for (int i = 0; i < 4; i++) attw[b * S_ + lane * 4 + i] = v[i] * inv;
    }
    grid_barrier(ctrl, NBLK);
    // ---- phase c: ctx[b][d] = sum_s attw[b][s] * h[s,b,d]
    for (int o = gtid; o < B_ * DH_; o += 16384) {
      int b = o >> 10, d = o & 1023;
      float acc = 0.f;
      for (int s = 0; s < S_; s++)
        acc += attw[b * S_ + s] * bf2f(hbf[(size_t)(s * B_ + b) * DH_ + d]);
      ctxBF[o] = f2bf(acc);
    }
    grid_barrier(ctrl, NBLK);
    // ---- phase d: gates[32,4096] = P[t] + ctx@WihCtx^T + state@Whh^T (LDS weights)
    {
      int mt = wv & 1, nt = wv >> 1;            // 2 M-tiles x 4 N-tiles per block
      int row0 = mt * 16;
      v8f acc = {};
      for (int kb = 0; kb < DH_; kb += 32) {
        v16bf bfr = load_bfrag_lds(sW, DH_, nt * 16, kb, lane);
        v16bf a0  = load_afrag(ctxBF, DH_, row0, kb, lane);
        acc = wmma_bf16(a0, bfr, acc);
      }
      for (int kb = 0; kb < DH_; kb += 32) {
        v16bf bfr = load_bfrag_lds(sW + 64 * 1024, DH_, nt * 16, kb, lane);
        v16bf a0  = load_afrag(stBF, DH_, row0, kb, lane);
        acc = wmma_bf16(a0, bfr, acc);
      }
      int col = colBase + nt * 16 + (lane & 15);
      int rb  = row0 + ((lane >> 4) * 8);
      const float* Prow = P + (size_t)(t * B_) * (4 * DH_);
#pragma unroll
      for (int r = 0; r < 8; r++) {
        int m = rb + r;
        gates[(size_t)m * 4096 + col] = acc[r] + Prow[(size_t)m * 4096 + col];
      }
    }
    grid_barrier(ctrl, NBLK);
    // ---- phase e: LSTM cell on [32,1024]
    for (int o = gtid; o < B_ * DH_; o += 16384) {
      int b = o >> 10, d = o & 1023;
      float gi = gates[b * 4096 + d];
      float gf = gates[b * 4096 + 1024 + d];
      float gg = gates[b * 4096 + 2048 + d];
      float go = gates[b * 4096 + 3072 + d];
      float c  = stC[o];
      float c2 = sigmoidf_(gf) * c + sigmoidf_(gi) * tanhf_(gg);
      float s2 = sigmoidf_(go) * tanhf_(c2);
      stC[o] = c2;
      stS[o] = s2;
      stBF[o] = f2bf(s2);
      ysF[(size_t)t * (B_ * DH_) + o] = s2;
      if (lvl == 0) ys0BF[(size_t)t * (B_ * DH_) + o] = f2bf(s2);
    }
    grid_barrier(ctrl, NBLK);
  }
}

// ---------------------------------------------------------------------------
// Output projection: out[l,s,b,tag] = <ys[l,s,b,:], out_W[tag,:]> + out_b[tag]
// N=11 < one WMMA tile -> scalar fp32 (trivial FLOPs).
// ---------------------------------------------------------------------------
__global__ void outproj_kernel(const float* __restrict__ ys,
                               const float* __restrict__ Wout,
                               const float* __restrict__ bout,
                               float* __restrict__ out) {
  int o = blockIdx.x * blockDim.x + threadIdx.x;
  int total = L_ * S_ * B_ * TAG_;
  if (o >= total) return;
  int tg = o % TAG_;
  int r  = o / TAG_;
  const float* y = ys + (size_t)r * DH_;
  const float* w = Wout + (size_t)tg * DH_;
  float acc = bout[tg];
  for (int k = 0; k < DH_; k += 4) {
    acc += y[k] * w[k] + y[k + 1] * w[k + 1] + y[k + 2] * w[k + 2] + y[k + 3] * w[k + 3];
  }
  out[o] = acc;
}

// ---------------------------------------------------------------------------
// Host launcher
// ---------------------------------------------------------------------------
extern "C" void kernel_launch(void* const* d_in, const int* in_sizes, int n_in,
                              void* d_out, int out_size, void* d_ws, size_t ws_size,
                              hipStream_t stream) {
  (void)in_sizes; (void)n_in; (void)out_size;
  if (ws_size < WS_NEED) return;

  const int*   seqs   = (const int*)d_in[0];
  const float* emb    = (const float*)d_in[2];
  const float* efWih  = (const float*)d_in[3];
  const float* efWhh  = (const float*)d_in[4];
  const float* efB    = (const float*)d_in[5];
  const float* ebWih  = (const float*)d_in[6];
  const float* ebWhh  = (const float*)d_in[7];
  const float* ebB    = (const float*)d_in[8];
  const float* dWih   = (const float*)d_in[9];
  const float* dWhh   = (const float*)d_in[10];
  const float* dB     = (const float*)d_in[11];
  const float* attnW  = (const float*)d_in[12];
  const float* outW   = (const float*)d_in[13];
  const float* outB   = (const float*)d_in[14];

  char* w = (char*)d_ws;
  unsigned* CTRL  = (unsigned*)(w + OFF_CTRL);
  float*  STS     = (float*)(w + OFF_STS);
  float*  STC     = (float*)(w + OFF_STC);
  bf16_t* STBF    = (bf16_t*)(w + OFF_STBF);
  float*  ENCS    = (float*)(w + OFF_ENCS);
  float*  ENCC    = (float*)(w + OFF_ENCC);
  bf16_t* ENCSBF  = (bf16_t*)(w + OFF_ENCSBF);
  float*  LOG     = (float*)(w + OFF_LOG);
  float*  ATT     = (float*)(w + OFF_ATT);
  bf16_t* CTXBF   = (bf16_t*)(w + OFF_CTX);
  float*  GATES   = (float*)(w + OFF_GATES);
  bf16_t* XE      = (bf16_t*)(w + OFF_XE);
  bf16_t* WFIH    = (bf16_t*)(w + OFF_WFIH);
  bf16_t* WFHH    = (bf16_t*)(w + OFF_WFHH);
  bf16_t* WBIH    = (bf16_t*)(w + OFF_WBIH);
  bf16_t* WBHH    = (bf16_t*)(w + OFF_WBHH);
  bf16_t* WDIH    = (bf16_t*)(w + OFF_WDIH);
  bf16_t* WDHH    = (bf16_t*)(w + OFF_WDHH);
  bf16_t* WATT    = (bf16_t*)(w + OFF_WATT);
  float*  GF      = (float*)(w + OFF_GF);
  float*  GB      = (float*)(w + OFF_GB);
  bf16_t* HBF     = (bf16_t*)(w + OFF_HBF);
  bf16_t* HW      = (bf16_t*)(w + OFF_HW);
  bf16_t* YS0     = (bf16_t*)(w + OFF_YS0);
  float*  YSF     = (float*)(w + OFF_YSF);
  float*  PBUF    = (float*)(w + OFF_P);

  // 1. zero barrier ctrl + all recurrent state
  {
    int nwords = (int)(OFF_ZEND / 4);
    zero_kernel<<<(nwords + 255) / 256, 256, 0, stream>>>((unsigned*)w, nwords);
  }
  // 2. weight fp32 -> bf16 conversions
  f2bf_kernel<<<1024, 256, 0, stream>>>(efWih, WFIH, 4 * H_ * E_);
  f2bf_kernel<<<1024, 256, 0, stream>>>(efWhh, WFHH, 4 * H_ * H_);
  f2bf_kernel<<<1024, 256, 0, stream>>>(ebWih, WBIH, 4 * H_ * E_);
  f2bf_kernel<<<1024, 256, 0, stream>>>(ebWhh, WBHH, 4 * H_ * H_);
  f2bf_kernel<<<2048, 256, 0, stream>>>(dWih,  WDIH, 4 * DH_ * 3 * DH_);
  f2bf_kernel<<<2048, 256, 0, stream>>>(dWhh,  WDHH, 4 * DH_ * DH_);
  f2bf_kernel<<<1024, 256, 0, stream>>>(attnW, WATT, DH_ * DH_);
  // 3. embedding gather -> bf16 [S*B, E]
  embed_kernel<<<S_ * B_, 128, 0, stream>>>(seqs, emb, XE);
  // 4/5. encoder input-gate precompute: G = x @ Wih^T + b  (f32 out)
  wmma_gemm_dual<<<dim3(4 * H_ / 128, S_ * B_ / 32), 256, 0, stream>>>(
      XE, WFIH, E_, E_, E_, nullptr, nullptr, 0, 0, 0, efB, GF, nullptr, 4 * H_);
  wmma_gemm_dual<<<dim3(4 * H_ / 128, S_ * B_ / 32), 256, 0, stream>>>(
      XE, WBIH, E_, E_, E_, nullptr, nullptr, 0, 0, 0, ebB, GB, nullptr, 4 * H_);
  // 6. persistent bidirectional encoder -> h bf16
  encoder_kernel<<<32, 256, 0, stream>>>(WFHH, WBHH, GF, GB, ENCS, ENCC, ENCSBF,
                                         GATES, HBF, CTRL);
  // 7. attention pre-projection hW = h @ attn_W^T  (bf16 out)
  wmma_gemm_dual<<<dim3(DH_ / 128, S_ * B_ / 32), 256, 0, stream>>>(
      HBF, WATT, DH_, DH_, DH_, nullptr, nullptr, 0, 0, 0, nullptr, nullptr, HW, DH_);
  // 8. decoder levels
  for (int lvl = 0; lvl < L_; lvl++) {
    const bf16_t* prev = (lvl == 0) ? HBF : YS0;
    // static decoder input gates: P = h@Wih[:,1024:2048]^T + prev@Wih[:,2048:3072]^T + b
    wmma_gemm_dual<<<dim3(4 * DH_ / 128, S_ * B_ / 32), 256, 0, stream>>>(
        HBF, WDIH + DH_, DH_, DH_, 3 * DH_,
        prev, WDIH + 2 * DH_, DH_, DH_, 3 * DH_,
        dB, PBUF, nullptr, 4 * DH_);
    decoder_kernel<<<64, 256, 0, stream>>>(
        HW, HBF, WDIH /* ctx cols, ld=3072 */, WDHH, PBUF, STS, STC, STBF,
        LOG, ATT, CTXBF, GATES, YSF + (size_t)lvl * S_ * B_ * DH_, YS0, lvl, CTRL);
  }
  // 9. output projection
  {
    int total = L_ * S_ * B_ * TAG_;
    outproj_kernel<<<(total + 255) / 256, 256, 0, stream>>>(YSF, outW, outB,
                                                            (float*)d_out);
  }
}
